// Multi_Head_Attention_90623809945792
// MI455X (gfx1250) — compile-verified, matrix operand order checked by reading, not run on hardware
//
#include <hip/hip_runtime.h>

// ---------------------------------------------------------------------------
// Multi-head attention forward for MI455X (gfx1250, wave32, WMMA).
// - All matmuls on v_wmma_f32_16x16x32_f16 (f16 in, f32 accumulate).
// - Flash-style online softmax with transposed score tiles (one shfl_xor(16)
//   per reduction; P -> B-fragment is a pure per-lane f32->f16 cvt).
// - Attention K/V tiles staged to LDS once per workgroup via CDNA5 async
//   global->LDS copies (ASYNCcnt), double-buffered to overlap with WMMA.
// ---------------------------------------------------------------------------

typedef _Float16 f16;
typedef __attribute__((ext_vector_type(16))) _Float16 v16h;
typedef __attribute__((ext_vector_type(8)))  _Float16 v8h;
typedef __attribute__((ext_vector_type(8)))  float    v8f;
typedef __attribute__((ext_vector_type(4)))  float    v4f;

#define WMMA_F16(a, b, c) \
  __builtin_amdgcn_wmma_f32_16x16x32_f16(false, (a), false, (b), (short)0, (c), false, false)

static constexpr int EMB = 1024;
static constexpr int SEQ = 2048;
static constexpr int NB  = 2;
static constexpr int NH  = 16;
static constexpr int DK  = 64;
static constexpr int MROWS = NB * SEQ;  // 4096

// f16 A/B fragment for 16x16x32: lane r=lane&15 is the row (A) / col (B),
// hl=lane>>4 selects the K-half. Element e<8 -> K = hl*8+e ; e>=8 -> K = hl*8+8+e.
// => two contiguous 8-half (16B) chunks at base and base+16.
__device__ inline v16h load_frag_f16(const f16* base) {
  union { v16h v; v8h h[2]; } u;
  u.h[0] = *(const v8h*)(base);
  u.h[1] = *(const v8h*)(base + 16);
  return u.v;
}

// Same fragment but sourced from fp32 memory (converted in-register).
__device__ inline v16h load_frag_f32(const float* base) {
  v4f c0 = *(const v4f*)(base);
  v4f c1 = *(const v4f*)(base + 4);
  v4f c2 = *(const v4f*)(base + 16);
  v4f c3 = *(const v4f*)(base + 20);
  v16h v;
#pragma unroll
  for (int i = 0; i < 4; i++) {
    v[i]      = (f16)c0[i];
    v[4 + i]  = (f16)c1[i];
    v[8 + i]  = (f16)c2[i];
    v[12 + i] = (f16)c3[i];
  }
  return v;
}

// CDNA5 async global->LDS copy of 16 bytes (VGLOBAL GLOBAL_LOAD_ASYNC_TO_LDS_B128,
// GV mode: vdst = LDS byte address, vaddr = 64-bit global address). ASYNCcnt.
__device__ __forceinline__ void async_copy16(f16* lds_dst, const f16* gsrc) {
  uint32_t lds_off = (uint32_t)(uintptr_t)lds_dst;  // aperture check keeps addr[31:0]
  asm volatile("global_load_async_to_lds_b128 %0, %1, off"
               :: "v"(lds_off), "v"(gsrc)
               : "memory");
}
__device__ __forceinline__ void wait_async0() {
  asm volatile("s_wait_asynccnt 0x0" ::: "memory");
}

// ---------------------------------------------------------------------------
// Wt[n*1024 + k] = (f16) W[k*1024 + n]   (weight transpose + f16 convert)
// ---------------------------------------------------------------------------
__global__ void k_wt_transpose(const float* __restrict__ W, f16* __restrict__ Wt) {
  int idx = blockIdx.x * blockDim.x + threadIdx.x;   // 0 .. 1M-1
  int n = idx >> 10;
  int k = idx & 1023;
  Wt[idx] = (f16)W[(size_t)k * EMB + n];
}

// ---------------------------------------------------------------------------
// Fused QKV projection.  z = blockIdx.z selects (input, weight, output layout):
//   z=0: Q = (query @ Wq) * 0.125 -> Qb[b,h,s,d]   (f16)
//   z=1: K =  key   @ Wk          -> Kb[b,h,s,d]   (f16)
//   z=2: V =  value @ Wv          -> Vt[b,h,d,s]   (f16, transposed per head)
// Wave tile 64x64, WG = 4 waves = 128x128 tile, K-loop 1024 in steps of 32.
// ---------------------------------------------------------------------------
__global__ void __launch_bounds__(128)
k_proj_qkv(const float* __restrict__ qin, const float* __restrict__ kin,
           const float* __restrict__ vin, const f16* __restrict__ wt,
           f16* __restrict__ Qb, f16* __restrict__ Kb, f16* __restrict__ Vt) {
  const int z = blockIdx.z;
  const float* X  = (z == 0) ? qin : (z == 1) ? kin : vin;
  const f16*   Wt = wt + (size_t)z * EMB * EMB;

  const int lane = threadIdx.x & 31, wave = threadIdx.x >> 5;
  const int r = lane & 15, hl = lane >> 4;
  const int m0 = blockIdx.x * 128 + (wave >> 1) * 64;
  const int n0 = blockIdx.y * 128 + (wave & 1) * 64;

  v8f acc[4][4] = {};

  for (int k0 = 0; k0 < EMB; k0 += 32) {
    v16h af[4], bf[4];
#pragma unroll
    for (int t = 0; t < 4; t++)
      af[t] = load_frag_f32(X + (size_t)(m0 + t * 16 + r) * EMB + k0 + hl * 8);
#pragma unroll
    for (int t = 0; t < 4; t++)
      bf[t] = load_frag_f16(Wt + (size_t)(n0 + t * 16 + r) * EMB + k0 + hl * 8);
#pragma unroll
    for (int i = 0; i < 4; i++)
#pragma unroll
      for (int j = 0; j < 4; j++)
        acc[i][j] = WMMA_F16(af[i], bf[j], acc[i][j]);
  }

#pragma unroll
  for (int ti = 0; ti < 4; ti++) {
    const int mbase = m0 + ti * 16 + hl * 8;     // 8 consecutive token rows
    const int b = mbase >> 11;                   // / 2048 (never straddles)
    const int s = mbase & 2047;
#pragma unroll
    for (int tj = 0; tj < 4; tj++) {
      const int n = n0 + tj * 16 + r;
      const int head = n >> 6, d = n & 63;
      if (z == 2) {
        v8h pk;
#pragma unroll
        for (int i = 0; i < 8; i++) pk[i] = (f16)acc[ti][tj][i];
        *(v8h*)(Vt + ((size_t)(b * NH + head) * DK + d) * SEQ + s) = pk;  // 16B
      } else {
        f16* dst = (z == 0) ? Qb : Kb;
        const float sc = (z == 0) ? 0.125f : 1.0f;   // 1/sqrt(64) folded into Q
#pragma unroll
        for (int i = 0; i < 8; i++)
          dst[((size_t)(b * NH + head) * SEQ + (s + i)) * DK + d] =
              (f16)(acc[ti][tj][i] * sc);
      }
    }
  }
}

// ---------------------------------------------------------------------------
// Flash attention with double-buffered async K/V staging into LDS.
// WG = 8 waves; each wave owns a 16-row q tile; all waves share the staged
// K (32x64) and V^T (64x32) tiles per 32-kv step (8 KB / step / WG).
// Rows padded (64->72, 32->40 halves) to spread ds_load_b128 across banks.
// ---------------------------------------------------------------------------
__global__ void __launch_bounds__(256)
k_attn(const f16* __restrict__ Qb, const f16* __restrict__ Kb,
       const f16* __restrict__ Vt, const unsigned char* __restrict__ mask,
       f16* __restrict__ ctx) {
  __shared__ f16 sK[2][32][72];   // [buf][kv][d]   (72 = 64 + 8 pad)
  __shared__ f16 sV[2][64][40];   // [buf][d][kv]   (40 = 32 + 8 pad)

  const int bh = blockIdx.y;                 // 0..31
  const int b = bh >> 4, head = bh & 15;
  const int t = threadIdx.x;
  const int lane = t & 31, wave = t >> 5;
  const int r = lane & 15, hl = lane >> 4;
  const int q0 = (blockIdx.x * 8 + wave) * 16;

  const f16* Qr = Qb + (size_t)bh * SEQ * DK;
  const f16* Kr = Kb + (size_t)bh * SEQ * DK;
  const f16* Vr = Vt + (size_t)bh * DK * SEQ;
  const unsigned char* mr = mask + ((size_t)b * SEQ + (q0 + r)) * SEQ;  // row q

  // Per-thread staging slots: K: 32 rows x 8 chunks, V: 64 rows x 4 chunks.
  const int krow = t >> 3, kc = (t & 7) * 8;   // K: 256 thr * 16B = 4KB
  const int vrow = t >> 2, vc = (t & 3) * 8;   // V: 256 thr * 16B = 4KB

  // Q^T B-fragments (resident for whole kv loop): col q = q0+r, K-dim = d.
  v16h qf[2];
#pragma unroll
  for (int tt = 0; tt < 2; tt++)
    qf[tt] = load_frag_f16(Qr + (size_t)(q0 + r) * DK + tt * 32 + hl * 8);

  v8f acc[4] = {};                 // O^T: 4 d-tiles x 8 VGPR (q = lane&15)
  float mrun = -1.0e30f, lrun = 0.0f;
  const float L2E = 1.44269504088896340736f;

  // Prologue: stage kv tile 0 into buffer 0.
  async_copy16(&sK[0][krow][kc], Kr + (size_t)krow * DK + kc);
  async_copy16(&sV[0][vrow][vc], Vr + (size_t)vrow * SEQ + vc);
  wait_async0();
  __syncthreads();

  for (int kv0 = 0; kv0 < SEQ; kv0 += 32) {
    const int cur = (kv0 >> 5) & 1;
    const bool has_next = (kv0 + 32) < SEQ;

    // Kick off next tile's async copies (other buffer) before computing.
    if (has_next) {
      const int nkv = kv0 + 32;
      async_copy16(&sK[cur ^ 1][krow][kc], Kr + (size_t)(nkv + krow) * DK + kc);
      async_copy16(&sV[cur ^ 1][vrow][vc], Vr + (size_t)vrow * SEQ + nkv + vc);
    }

    // --- scores: two 16(kv) x 16(q) tiles, transposed layout (from LDS) ---
    v8f s0 = {}, s1 = {};
    {
      v16h ka = load_frag_f16(&sK[cur][r][hl * 8]);
      v16h kc2 = load_frag_f16(&sK[cur][r][32 + hl * 8]);
      s0 = WMMA_F16(ka, qf[0], s0);
      s0 = WMMA_F16(kc2, qf[1], s0);
    }
    {
      v16h ka = load_frag_f16(&sK[cur][16 + r][hl * 8]);
      v16h kc2 = load_frag_f16(&sK[cur][16 + r][32 + hl * 8]);
      s1 = WMMA_F16(ka, qf[0], s1);
      s1 = WMMA_F16(kc2, qf[1], s1);
    }

    // --- mask: lane holds kv = kv0 + t16*16 + hl*8 + i for fixed q ---
    uint2 mv0 = *(const uint2*)(mr + kv0 + hl * 8);
    uint2 mv1 = *(const uint2*)(mr + kv0 + 16 + hl * 8);
#pragma unroll
    for (int i = 0; i < 4; i++) {
      if ((mv0.x >> (8 * i)) & 0xff) s0[i]     = -1.0e9f;
      if ((mv0.y >> (8 * i)) & 0xff) s0[4 + i] = -1.0e9f;
      if ((mv1.x >> (8 * i)) & 0xff) s1[i]     = -1.0e9f;
      if ((mv1.y >> (8 * i)) & 0xff) s1[4 + i] = -1.0e9f;
    }

    // --- online softmax per q column (lane pair {lane, lane^16}) ---
    float cmax = s0[0];
#pragma unroll
    for (int i = 1; i < 8; i++) cmax = fmaxf(cmax, s0[i]);
#pragma unroll
    for (int i = 0; i < 8; i++) cmax = fmaxf(cmax, s1[i]);
    cmax = fmaxf(cmax, __shfl_xor(cmax, 16, 32));
    const float mnew = fmaxf(mrun, cmax);
    const float rescale = __builtin_exp2f((mrun - mnew) * L2E);

    float psum = 0.0f;
    v16h p;   // P^T B-fragment: C-layout of s0/s1 matches element order exactly
#pragma unroll
    for (int i = 0; i < 8; i++) {
      float pv = __builtin_exp2f((s0[i] - mnew) * L2E);
      psum += pv; p[i] = (f16)pv;
    }
#pragma unroll
    for (int i = 0; i < 8; i++) {
      float pv = __builtin_exp2f((s1[i] - mnew) * L2E);
      psum += pv; p[8 + i] = (f16)pv;
    }
    psum += __shfl_xor(psum, 16, 32);
    lrun = lrun * rescale + psum;
    mrun = mnew;

#pragma unroll
    for (int dt = 0; dt < 4; dt++)
#pragma unroll
      for (int i = 0; i < 8; i++) acc[dt][i] *= rescale;

    // --- O^T += V^T x P^T : A rows = d, K-dim = kv (from LDS) ---
#pragma unroll
    for (int dt = 0; dt < 4; dt++) {
      v16h vf = load_frag_f16(&sV[cur][dt * 16 + r][hl * 8]);
      acc[dt] = WMMA_F16(vf, p, acc[dt]);
    }

    // Next tile's async copies must land before anyone reads them.
    if (has_next) {
      wait_async0();
      __syncthreads();
    }
  }

  // --- normalize + store context [b, s, h*64+d] as f16 (16B per d-tile) ---
  const float inv = 1.0f / lrun;
  f16* crow = ctx + ((size_t)b * SEQ + q0 + r) * EMB + head * DK;
#pragma unroll
  for (int dt = 0; dt < 4; dt++) {
    v8h pk;
#pragma unroll
    for (int i = 0; i < 8; i++) pk[i] = (f16)(acc[dt][i] * inv);
    *(v8h*)(crow + dt * 16 + hl * 8) = pk;
  }
}

// ---------------------------------------------------------------------------
// Output projection: out(f32) = ctx(f16) @ Wo  (Wo^T f16 in workspace).
// ---------------------------------------------------------------------------
__global__ void __launch_bounds__(128)
k_proj_out(const f16* __restrict__ ctx, const f16* __restrict__ Wot,
           float* __restrict__ out) {
  const int lane = threadIdx.x & 31, wave = threadIdx.x >> 5;
  const int r = lane & 15, hl = lane >> 4;
  const int m0 = blockIdx.x * 128 + (wave >> 1) * 64;
  const int n0 = blockIdx.y * 128 + (wave & 1) * 64;

  v8f acc[4][4] = {};
  for (int k0 = 0; k0 < EMB; k0 += 32) {
    v16h af[4], bf[4];
#pragma unroll
    for (int t = 0; t < 4; t++)
      af[t] = load_frag_f16(ctx + (size_t)(m0 + t * 16 + r) * EMB + k0 + hl * 8);
#pragma unroll
    for (int t = 0; t < 4; t++)
      bf[t] = load_frag_f16(Wot + (size_t)(n0 + t * 16 + r) * EMB + k0 + hl * 8);
#pragma unroll
    for (int i = 0; i < 4; i++)
#pragma unroll
      for (int j = 0; j < 4; j++)
        acc[i][j] = WMMA_F16(af[i], bf[j], acc[i][j]);
  }

#pragma unroll
  for (int ti = 0; ti < 4; ti++)
#pragma unroll
    for (int tj = 0; tj < 4; tj++) {
      const int n = n0 + tj * 16 + r;
#pragma unroll
      for (int i = 0; i < 8; i++) {
        const int m = m0 + ti * 16 + hl * 8 + i;
        out[(size_t)m * EMB + n] = acc[ti][tj][i];
      }
    }
}

// ---------------------------------------------------------------------------
extern "C" void kernel_launch(void* const* d_in, const int* in_sizes, int n_in,
                              void* d_out, int out_size, void* d_ws, size_t ws_size,
                              hipStream_t stream) {
  (void)in_sizes; (void)n_in; (void)out_size; (void)ws_size;
  const float* qin = (const float*)d_in[0];
  const float* kin = (const float*)d_in[1];
  const float* vin = (const float*)d_in[2];
  const unsigned char* mask = (const unsigned char*)d_in[3];  // jax bool_: 1 byte
  const float* Wq = (const float*)d_in[4];
  const float* Wk = (const float*)d_in[5];
  const float* Wv = (const float*)d_in[6];
  const float* Wo = (const float*)d_in[7];
  float* out = (float*)d_out;

  char* ws = (char*)d_ws;
  const size_t MB = 1024u * 1024u;
  const size_t WSZ = (size_t)EMB * EMB;          // 1M f16 elems per weight
  f16* wt  = (f16*)(ws);                         // 4 weights^T : 8 MB
  f16* qb  = (f16*)(ws + 8 * MB);                // Q  [B,H,S,DK] : 8 MB
  f16* kb  = (f16*)(ws + 16 * MB);               // K  [B,H,S,DK] : 8 MB
  f16* vtb = (f16*)(ws + 24 * MB);               // V^T[B,H,DK,S] : 8 MB
  f16* ctx = (f16*)(ws + 32 * MB);               // context [B,S,E] : 8 MB

  // 1. weights -> f16, transposed (rows = output feature, contiguous K)
  k_wt_transpose<<<4096, 256, 0, stream>>>(Wq, wt + 0 * WSZ);
  k_wt_transpose<<<4096, 256, 0, stream>>>(Wk, wt + 1 * WSZ);
  k_wt_transpose<<<4096, 256, 0, stream>>>(Wv, wt + 2 * WSZ);
  k_wt_transpose<<<4096, 256, 0, stream>>>(Wo, wt + 3 * WSZ);

  // 2. fused QKV projections (z = 0/1/2)
  dim3 gp(MROWS / 128, EMB / 128, 3);
  k_proj_qkv<<<gp, 128, 0, stream>>>(qin, kin, vin, wt, qb, kb, vtb);

  // 3. flash attention (8 q-tiles of 16 rows per WG of 8 waves)
  dim3 ga(SEQ / (16 * 8), NB * NH);
  k_attn<<<ga, 256, 0, stream>>>(qb, kb, vtb, mask, ctx);

  // 4. output projection -> f32
  dim3 go(MROWS / 128, EMB / 128);
  k_proj_out<<<go, 128, 0, stream>>>(ctx, wt + 3 * WSZ, out);
}
